// SparseNet_40037685133497
// MI455X (gfx1250) — matrix operand order repaired and checked by reading, not compiled, not run on hardware
//
#include <hip/hip_runtime.h>

typedef __attribute__((ext_vector_type(16))) _Float16 v16h;
typedef __attribute__((ext_vector_type(8)))  float    v8f;

#define NB      64          // batch
#define D1      30          // conv1 output spatial
#define D2      28          // conv2 output spatial
#define C1      32          // conv1 out channels
#define C2      64          // conv2 out channels
#define MTOT    (NB*D2*D2*D2)          // 1,404,928 GEMM rows
#define N1TOT   (NB*D1*D1*D1)          // conv1 voxels
#define H1_STRIDE_C 32

// workspace layout
#define WS_BT_OFF  0                    // 27*64*32 f16 = 110,592 B
#define WS_H1_OFF  131072               // h1: N1TOT*32 f16 = 110,592,000 B

// ---------------------------------------------------------------------------
// Kernel 0: convert + transpose conv2 weights: w2[kd][kh][kw][ci][co] (f32)
//           -> Bt[tap][co][ci] (f16), N-major per tap for conflict-free
//           WMMA B-fragment reads.
// ---------------------------------------------------------------------------
__global__ void prep_w2(const float* __restrict__ w2, _Float16* __restrict__ Bt) {
    int t = blockIdx.x * 256 + threadIdx.x;
    if (t >= 27 * 64 * 32) return;
    int tap = t >> 11;          // /2048
    int rem = t & 2047;
    int co  = rem >> 5;
    int ci  = rem & 31;
    Bt[t] = (_Float16)w2[(tap * 32 + ci) * 64 + co];   // Bt flat == tap*2048+co*32+ci
}

// ---------------------------------------------------------------------------
// Kernel 1: conv1 (K=27, Cin=1, Cout=32) direct f32, output f16 channels-last.
// x fits in L2 (8.4 MB); only ~3 GFLOP -> VALU is plenty.
// ---------------------------------------------------------------------------
__global__ __launch_bounds__(256)
void conv1_kernel(const float* __restrict__ x, const float* __restrict__ w1,
                  const float* __restrict__ b1, _Float16* __restrict__ h1) {
    __shared__ float w1s[27 * 32];
    __shared__ float b1s[32];
    for (int i = threadIdx.x; i < 27 * 32; i += 256) w1s[i] = w1[i];
    if (threadIdx.x < 32) b1s[threadIdx.x] = b1[threadIdx.x];
    __syncthreads();

    int idx = blockIdx.x * 256 + threadIdx.x;
    if (idx >= N1TOT) return;
    int xo = idx % D1;
    int t1 = idx / D1;
    int yo = t1 % D1;
    int t2 = t1 / D1;
    int zo = t2 % D1;
    int b  = t2 / D1;

    float acc[32];
#pragma unroll
    for (int c = 0; c < 32; ++c) acc[c] = b1s[c];

    const float* xb = x + ((b * 32 + zo) * 32 + yo) * 32 + xo;
    for (int tap = 0; tap < 27; ++tap) {
        int kd = tap / 9, kh = (tap / 3) % 3, kw = tap % 3;
        float xv = xb[(kd * 32 + kh) * 32 + kw];
        const float* wrow = &w1s[tap * 32];
#pragma unroll
        for (int c = 0; c < 32; ++c) acc[c] += xv * wrow[c];
    }

    union { _Float16 h[32]; uint4 q[4]; } o;
#pragma unroll
    for (int c = 0; c < 32; ++c) {
        float v = acc[c];
        o.h[c] = (_Float16)(v > 0.f ? v : 0.f);
    }
    uint4* dst = (uint4*)(h1 + (size_t)idx * 32);
    dst[0] = o.q[0]; dst[1] = o.q[1]; dst[2] = o.q[2]; dst[3] = o.q[3];
}

// ---------------------------------------------------------------------------
// Kernel 2: conv2 as implicit GEMM.  M = 64*28^3 voxels, N = 64,
// K = 27 taps x 32 channels.  Block = 256 thr (8 waves), tile 128(M) x 64(N).
// Double-buffered LDS staging: one barrier per tap; next-tap global loads are
// issued before the WMMAs, staged ds_stores after (DS is in-order per wave,
// so fragment ds_loads must precede staging ds_stores in program order).
// Epilogue transposes through LDS for coalesced NCDHW stores.
// ---------------------------------------------------------------------------
__global__ __launch_bounds__(256)
void conv2_kernel(const _Float16* __restrict__ h1, const _Float16* __restrict__ Bt,
                  const float* __restrict__ b2, float* __restrict__ out) {
    union alignas(16) SM {
        struct {
            _Float16 A[2][128][40];   // M x K tiles, padded row stride (80 B)
            _Float16 B[2][64][40];    // N-major (transposed) K tiles
        } s;
        float D[128][65];             // f32 result tile for coalesced epilogue
    };
    __shared__ SM sm;

    const int t    = threadIdx.x;
    const int lane = t & 31;
    const int wv   = t >> 5;
    const int tileBase = blockIdx.x * 128;

    // ---- per-thread A-staging coords: voxel i = t>>1, 16-channel half = t&1
    const int ai  = t >> 1;
    const int ac0 = (t & 1) * 16;
    const int mg  = tileBase + ai;
    const bool avalid = (mg < MTOT);
    const int mc = avalid ? mg : 0;
    const int vx = mc % D2;
    const int r1 = mc / D2;
    const int vy = r1 % D2;
    const int r2 = r1 / D2;
    const int vz = r2 % D2;
    const int vb = r2 / D2;
    const _Float16* abase =
        h1 + (((vb * D1 + vz) * D1 + vy) * D1 + vx) * H1_STRIDE_C + ac0;

    // ---- B-staging coords: Bt[tap] is contiguous [64][32] f16 (4 KB)
    const int bi = t >> 2;              // n 0..63
    const int bj = (t & 3) << 3;        // k 0,8,16,24

    // ---- WMMA fragment LDS addressing (contiguous k = 16*(lane/16) + s for A and B, wave32)
    const int arow = wv * 16 + (lane & 15);     // A: M row
    const int akb  = (lane >> 4) << 4;          // A: K base 0 or 16
    const int bn   = lane & 15;                 // B: N (within 16)
    const int bk0  = (lane >> 4) << 4;          // B: K base 0 or 16

    v8f acc0 = {}, acc1 = {}, acc2 = {}, acc3 = {};

    // ---- prologue: stage tap 0 into buffer 0
    {
        uint4 q0 = make_uint4(0, 0, 0, 0), q1 = q0;
        if (avalid) {
            const uint4* s4 = (const uint4*)abase;   // tap 0 = offset 0
            q0 = s4[0];
            q1 = s4[1];
        }
        const uint4 bq = ((const uint4*)Bt)[t];
        *(uint4*)&sm.s.A[0][ai][ac0]     = q0;
        *(uint4*)&sm.s.A[0][ai][ac0 + 8] = q1;
        *(uint4*)&sm.s.B[0][bi][bj]      = bq;
    }

    for (int tap = 0; tap < 27; ++tap) {
        __syncthreads();
        const int cur = tap & 1;

        // ---- issue next-tap global loads into registers (latency hidden
        //      behind the WMMAs below)
        uint4 q0 = make_uint4(0, 0, 0, 0), q1 = q0, bq = q0;
        if (tap < 26) {
            const int nt  = tap + 1;
            const int nkd = nt / 9, nkh = (nt / 3) % 3, nkw = nt % 3;
            if (avalid) {
                const uint4* s4 = (const uint4*)(
                    abase + ((nkd * D1 + nkh) * D1 + nkw) * H1_STRIDE_C);
                q0 = s4[0];
                q1 = s4[1];
                if (tap < 25) {
                    const int pt  = tap + 2;
                    const int pkd = pt / 9, pkh = (pt / 3) % 3, pkw = pt % 3;
                    __builtin_prefetch(
                        abase + ((pkd * D1 + pkh) * D1 + pkw) * H1_STRIDE_C, 0, 0);
                }
            }
            bq = ((const uint4*)(Bt + nt * 2048))[t];
        }

        // ---- fragments + 4x WMMA (N = 0,16,32,48) from current buffer
        union { uint4 q[2]; v16h v; } af, bf;
        af.q[0] = *(const uint4*)&sm.s.A[cur][arow][akb];       // K=akb..akb+7
        af.q[1] = *(const uint4*)&sm.s.A[cur][arow][akb + 8];   // K=akb+8..+15
        const v16h a = af.v;

        bf.q[0] = *(const uint4*)&sm.s.B[cur][bn][bk0];
        bf.q[1] = *(const uint4*)&sm.s.B[cur][bn][bk0 + 8];
        acc0 = __builtin_amdgcn_wmma_f32_16x16x32_f16(false, a, false, bf.v,
                                                      (short)0, acc0, false, false);
        bf.q[0] = *(const uint4*)&sm.s.B[cur][bn + 16][bk0];
        bf.q[1] = *(const uint4*)&sm.s.B[cur][bn + 16][bk0 + 8];
        acc1 = __builtin_amdgcn_wmma_f32_16x16x32_f16(false, a, false, bf.v,
                                                      (short)0, acc1, false, false);
        bf.q[0] = *(const uint4*)&sm.s.B[cur][bn + 32][bk0];
        bf.q[1] = *(const uint4*)&sm.s.B[cur][bn + 32][bk0 + 8];
        acc2 = __builtin_amdgcn_wmma_f32_16x16x32_f16(false, a, false, bf.v,
                                                      (short)0, acc2, false, false);
        bf.q[0] = *(const uint4*)&sm.s.B[cur][bn + 48][bk0];
        bf.q[1] = *(const uint4*)&sm.s.B[cur][bn + 48][bk0 + 8];
        acc3 = __builtin_amdgcn_wmma_f32_16x16x32_f16(false, a, false, bf.v,
                                                      (short)0, acc3, false, false);

        // ---- store staged registers into the other buffer (read next tap)
        if (tap < 26) {
            const int nxt = cur ^ 1;
            *(uint4*)&sm.s.A[nxt][ai][ac0]     = q0;
            *(uint4*)&sm.s.A[nxt][ai][ac0 + 8] = q1;
            *(uint4*)&sm.s.B[nxt][bi][bj]      = bq;
        }
    }
    __syncthreads();   // all fragment reads done before D overwrites the union

    // ---- epilogue: D tile -> LDS (f32), then coalesced NCDHW stores
    {
        const int mbase = wv * 16 + ((lane >> 4) << 3);   // + r gives M row
        const int ncol  = lane & 15;
#pragma unroll
        for (int r = 0; r < 8; ++r) {
            sm.D[mbase + r][ncol]      = acc0[r];
            sm.D[mbase + r][ncol + 16] = acc1[r];
            sm.D[mbase + r][ncol + 32] = acc2[r];
            sm.D[mbase + r][ncol + 48] = acc3[r];
        }
    }
    __syncthreads();

    for (int it = 0; it < 32; ++it) {
        const int idx = it * 256 + t;
        const int n   = idx >> 7;       // 0..63
        const int ml  = idx & 127;
        const int mg2 = tileBase + ml;
        if (mg2 < MTOT) {
            const int ox = mg2 % D2;
            const int s1 = mg2 / D2;
            const int oy = s1 % D2;
            const int s2 = s1 / D2;
            const int oz = s2 % D2;
            const int ob = s2 / D2;
            float v = sm.D[ml][n] + b2[n];
            v = v > 0.f ? v : 0.f;
            out[(((ob * C2 + n) * D2 + oz) * D2 + oy) * D2 + ox] = v;
        }
    }
}

// ---------------------------------------------------------------------------
extern "C" void kernel_launch(void* const* d_in, const int* in_sizes, int n_in,
                              void* d_out, int out_size, void* d_ws, size_t ws_size,
                              hipStream_t stream) {
    const float* x  = (const float*)d_in[0];
    const float* w1 = (const float*)d_in[1];
    const float* b1 = (const float*)d_in[2];
    const float* w2 = (const float*)d_in[3];
    const float* b2 = (const float*)d_in[4];
    float* out = (float*)d_out;

    _Float16* Bt = (_Float16*)((char*)d_ws + WS_BT_OFF);
    _Float16* h1 = (_Float16*)((char*)d_ws + WS_H1_OFF);

    // 0) weight convert/transpose: 27*64*32 elems
    prep_w2<<<(27 * 64 * 32 + 255) / 256, 256, 0, stream>>>(w2, Bt);

    // 1) conv1 -> f16 intermediate (channels-last)
    conv1_kernel<<<(N1TOT + 255) / 256, 256, 0, stream>>>(x, w1, b1, h1);

    // 2) conv2 implicit GEMM with WMMA, fused bias+ReLU+NCDHW transpose
    conv2_kernel<<<(MTOT + 127) / 128, 256, 0, stream>>>(h1, Bt, b2, out);
}
